// AttModule_mamba_cross_4088808865963
// MI455X (gfx1250) — matrix operand order repaired and checked by reading, not compiled, not run on hardware
//
#include <hip/hip_runtime.h>
#include <hip/hip_bf16.h>

// ---------------------------------------------------------------------------
// Shapes (compile-time constants from the reference)
// ---------------------------------------------------------------------------
#define Bn 4
#define Dn 256
#define Ln 2048
#define En 512
#define Nn 16
#define Kn 7
#define Rn 16

typedef __bf16 bf16x16 __attribute__((ext_vector_type(16)));
typedef __bf16 bf16x8  __attribute__((ext_vector_type(8)));
typedef float  f32x8   __attribute__((ext_vector_type(8)));

// CDNA5 WMMA: D(16x16 f32) = A(16x32 bf16) x B(32x16 bf16) + C
__device__ __forceinline__ f32x8 wmma_bf16(bf16x16 a, bf16x16 b, f32x8 c) {
    return __builtin_amdgcn_wmma_f32_16x16x32_bf16(
        /*neg_a=*/false, a, /*neg_b=*/false, b,
        /*c_mod=*/(short)0, c, /*reuse_a=*/false, /*reuse_b=*/false);
}

// A fragment (16x32 bf16), K-contiguous source: A[m][k] = src[m*strideM + k].
// lane l holds row m = l&15; kb = (l>>4)*8.
// elems 0..7 -> K=kb..kb+7, elems 8..15 -> K=kb+16..kb+23.
__device__ __forceinline__ bf16x16 load_a_bf(const __bf16* __restrict__ src,
                                             int strideM, int lane) {
    const int m  = lane & 15;
    const int kb = (lane >> 4) << 3;
    const __bf16* p = src + m * strideM + kb;
    bf16x8 lo = *(const bf16x8*)(p);
    bf16x8 hi = *(const bf16x8*)(p + 16);
    bf16x16 a;
#pragma unroll
    for (int i = 0; i < 8; ++i) { a[i] = lo[i]; a[i + 8] = hi[i]; }
    return a;
}

// B fragment (32x16 bf16), K-contiguous source: B[k][n] = src[n*strideN + k].
// lane l holds col n = l&15; kb = (l>>4)*8; elems 0..7 -> K=kb..kb+7, elems 8..15 -> K=kb+16..kb+23.
__device__ __forceinline__ bf16x16 load_b_bf(const __bf16* __restrict__ src,
                                             int strideN, int lane) {
    const int n  = lane & 15;
    const int kb = (lane >> 4) << 3;
    const __bf16* p = src + n * strideN + kb;
    bf16x8 lo = *(const bf16x8*)(p);
    bf16x8 hi = *(const bf16x8*)(p + 16);
    bf16x16 r;
#pragma unroll
    for (int i = 0; i < 8; ++i) { r[i] = lo[i]; r[i + 8] = hi[i]; }
    return r;
}

__device__ __forceinline__ float siluf(float v) {
    return v / (1.f + __expf(-v));
}

// ---------------------------------------------------------------------------
// Cast kernels: fp32 -> bf16 (native v_cvt, RNE)
// ---------------------------------------------------------------------------
__global__ void cast_kernel(const float* __restrict__ s, __bf16* __restrict__ d,
                            int n) {
    int i = blockIdx.x * 256 + threadIdx.x;
    if (i < n) d[i] = (__bf16)s[i];
}

// ff_w (D,D,K) -> bf16 transposed [k][d][c] so conv B-fragments are K(=c)-contig
__global__ void cast_ffw_kernel(const float* __restrict__ w,
                                __bf16* __restrict__ wT) {
    int idx = blockIdx.x * 256 + threadIdx.x;          // over D*D*K
    if (idx >= Dn * Dn * Kn) return;
    int k = idx % Kn;
    int c = (idx / Kn) % Dn;
    int d = idx / (Kn * Dn);
    wT[((size_t)k * Dn + d) * Dn + c] = (__bf16)w[idx];
}

// ---------------------------------------------------------------------------
// Kernel 1: conv_ff as implicit GEMM. out0[b,d,l] = relu(b + sum_{c,k} x*w)
// x tile staged in LDS as bf16 (position-major), 7 WMMAs per 32-channel block.
// Interior tiles take an unconditional vector-load fast path; edge tiles are
// branchless (clamped address + cndmask select of zero).
// ---------------------------------------------------------------------------
__global__ void conv_ff_kernel(const float* __restrict__ x,
                               const __bf16* __restrict__ wT,   // [k][d][c]
                               const float* __restrict__ bias,
                               float* __restrict__ out0) {
    const int lane = threadIdx.x;
    const int l0 = blockIdx.x * 16;
    const int d0 = blockIdx.y * 16;
    const int b  = blockIdx.z;

    __shared__ __align__(32) __bf16 lds_x[22 * 32];   // [position 0..21][channel 0..31]

    const bool interior = (l0 >= 3) && (l0 + 19 <= Ln);   // uniform per block

    f32x8 c = {};
    for (int cb = 0; cb < Dn; cb += 32) {
        // stage x[b, cb+lane, l0-3 .. l0+18] into LDS, bf16, position-major
        const float* xrow = x + ((size_t)b * Dn + cb + lane) * Ln;
        if (interior) {
            const float* p = xrow + (l0 - 3);
#pragma unroll
            for (int j = 0; j < 22; ++j)
                lds_x[j * 32 + lane] = (__bf16)p[j];
        } else {
#pragma unroll
            for (int j = 0; j < 22; ++j) {
                int l  = l0 - 3 + j;
                int lc = min(max(l, 0), Ln - 1);
                float v = xrow[lc];
                lds_x[j * 32 + lane] = (l == lc) ? (__bf16)v : (__bf16)0.f;
            }
        }
        __syncthreads();
#pragma unroll
        for (int k = 0; k < Kn; ++k) {
            // A[m][kk] = lds_x[(m+k)*32 + kk]   (K-contiguous)
            bf16x16 a  = load_a_bf(lds_x + k * 32, 32, lane);
            // B[kk][n] = wT[(k*D + d0+n)*D + cb+kk]   (K-contiguous)
            bf16x16 bm = load_b_bf(wT + ((size_t)k * Dn + d0) * Dn + cb, Dn, lane);
            c = wmma_bf16(a, bm, c);
        }
        __syncthreads();
    }
    const int n  = lane & 15;
    const int mb = (lane >> 4) << 3;
    const float bv = bias[d0 + n];
#pragma unroll
    for (int v = 0; v < 8; ++v) {
        int m = mb + v;
        float val = c[v] + bv;
        val = val > 0.f ? val : 0.f;
        out0[((size_t)b * Dn + d0 + n) * Ln + l0 + m] = val;
    }
}

// ---------------------------------------------------------------------------
// Kernel 2: instance norm over L per (b,d) row, fused mask-mul, writes bf16
// TRANSPOSED output xnT[b,l,d] so in_proj A-fragments are K(=d)-contiguous.
// ---------------------------------------------------------------------------
__global__ void instnorm_kernel(const float* __restrict__ src,
                                const float* __restrict__ mask,
                                __bf16* __restrict__ xnT) {
    const int row = blockIdx.x;           // b*D + d
    const int b   = row / Dn;
    const int dch = row % Dn;
    const int tid = threadIdx.x;
    const float* s = src + (size_t)row * Ln;

    __shared__ float r1[256], r2[256];
    float sum = 0.f, sq = 0.f;
    for (int i = tid; i < Ln; i += 256) {
        float v = s[i];
        sum += v; sq += v * v;
    }
    r1[tid] = sum; r2[tid] = sq;
    __syncthreads();
    for (int st = 128; st > 0; st >>= 1) {
        if (tid < st) { r1[tid] += r1[tid + st]; r2[tid] += r2[tid + st]; }
        __syncthreads();
    }
    const float mu  = r1[0] * (1.f / Ln);
    const float var = r2[0] * (1.f / Ln) - mu * mu;
    const float rs  = rsqrtf(var + 1e-5f);
    const float* mrow = mask + (size_t)b * Ln;
    for (int i = tid; i < Ln; i += 256)
        xnT[((size_t)b * Ln + i) * Dn + dch] = (__bf16)((s[i] - mu) * rs * mrow[i]);
}

// ---------------------------------------------------------------------------
// Kernel 3: in_proj GEMM  xz[b,l,e] = sum_d xnT[b,l,d] * w[e,d]   (e < 2E)
// ---------------------------------------------------------------------------
__global__ void inproj_kernel(const __bf16* __restrict__ xnT,
                              const __bf16* __restrict__ w,     // bf16 [2E][D]
                              float* __restrict__ xz) {
    const int lane = threadIdx.x;
    const int l0 = blockIdx.x * 16;
    const int e0 = blockIdx.y * 16;
    const int b  = blockIdx.z;

    f32x8 c = {};
    const __bf16* abase = xnT + ((size_t)b * Ln + l0) * Dn;  // A[m][k]=abase[m*D+k]
    const __bf16* bbase = w + (size_t)e0 * Dn;               // B[k][n]=bbase[n*D+k]
    for (int kb = 0; kb < Dn; kb += 32) {
        if (kb + 32 < Dn) __builtin_prefetch(abase + kb + 32, 0, 3);
        bf16x16 a  = load_a_bf(abase + kb, Dn, lane);
        bf16x16 bm = load_b_bf(bbase + kb, Dn, lane);
        c = wmma_bf16(a, bm, c);
    }
    const int n  = lane & 15;
    const int mb = (lane >> 4) << 3;
#pragma unroll
    for (int v = 0; v < 8; ++v) {
        int m = mb + v;
        xz[((size_t)b * Ln + l0 + m) * (2 * En) + e0 + n] = c[v];
    }
}

// ---------------------------------------------------------------------------
// Kernel 4: depthwise causal conv (K=7, left-pad 6) + SiLU on xp = xz[:,:, :E]
// Writes f32 (for scan) and bf16 copy (for x_proj A-fragments).
// ---------------------------------------------------------------------------
__global__ void dwconv_silu_kernel(const float* __restrict__ xz,
                                   const float* __restrict__ cw,
                                   const float* __restrict__ cb,
                                   float* __restrict__ xc,
                                   __bf16* __restrict__ xc_bf) {
    const size_t idx = (size_t)blockIdx.x * 256 + threadIdx.x;  // over B*L*E
    if (idx >= (size_t)Bn * Ln * En) return;
    const int e  = (int)(idx % En);
    const size_t bl = idx / En;
    const int l  = (int)(bl % Ln);
    const int b  = (int)(bl / Ln);
    float acc = cb[e];
#pragma unroll
    for (int k = 0; k < Kn; ++k) {
        int ls = l - 6 + k;
        if (ls >= 0)
            acc += xz[((size_t)b * Ln + ls) * (2 * En) + e] * cw[e * Kn + k];
    }
    float v = siluf(acc);
    xc[idx]    = v;
    xc_bf[idx] = (__bf16)v;
}

// ---------------------------------------------------------------------------
// Kernel 5: x_proj GEMM  xdbl[b,l,r] = sum_e xc[b,l,e] * w[r,e]   (r < 48)
// ---------------------------------------------------------------------------
__global__ void xproj_kernel(const __bf16* __restrict__ xc_bf,
                             const __bf16* __restrict__ w,       // bf16 [48][E]
                             float* __restrict__ xdbl) {
    const int lane = threadIdx.x;
    const int l0 = blockIdx.x * 16;
    const int r0 = blockIdx.y * 16;
    const int b  = blockIdx.z;

    f32x8 c = {};
    const __bf16* abase = xc_bf + ((size_t)b * Ln + l0) * En;
    const __bf16* bbase = w + (size_t)r0 * En;
    for (int kb = 0; kb < En; kb += 32) {
        if (kb + 32 < En) __builtin_prefetch(abase + kb + 32, 0, 3);
        bf16x16 a  = load_a_bf(abase + kb, En, lane);
        bf16x16 bm = load_b_bf(bbase + kb, En, lane);
        c = wmma_bf16(a, bm, c);
    }
    const int n  = lane & 15;
    const int mb = (lane >> 4) << 3;
#pragma unroll
    for (int v = 0; v < 8; ++v) {
        int m = mb + v;
        xdbl[((size_t)b * Ln + l0 + m) * 48 + r0 + n] = c[v];
    }
}

// ---------------------------------------------------------------------------
// Kernel 6: dt[b,l,e] = softplus(sum_r xdbl[b,l,r] * dtw[e,r] + dtb[e])
// ---------------------------------------------------------------------------
__global__ void dt_kernel(const float* __restrict__ xdbl,
                          const float* __restrict__ dtw,
                          const float* __restrict__ dtb,
                          float* __restrict__ dt) {
    const size_t idx = (size_t)blockIdx.x * 256 + threadIdx.x;  // over B*L*E
    if (idx >= (size_t)Bn * Ln * En) return;
    const int e = (int)(idx % En);
    const size_t bl = idx / En;
    const float* xr = xdbl + bl * 48;
    float acc = dtb[e];
#pragma unroll
    for (int r = 0; r < Rn; ++r) acc += xr[r] * dtw[e * Rn + r];
    dt[idx] = (acc > 20.f) ? acc : log1pf(__expf(acc));
}

// ---------------------------------------------------------------------------
// Kernel 7: selective scan, fused y *= silu(z); writes bf16 (out_proj operand)
// One thread per (b,e); 16-wide state in registers; B/C staged in LDS per l.
// ---------------------------------------------------------------------------
__global__ void scan_kernel(const float* __restrict__ xc,
                            const float* __restrict__ dt,
                            const float* __restrict__ xdbl,
                            const float* __restrict__ xz,
                            const float* __restrict__ A_log,
                            const float* __restrict__ Dp,
                            __bf16* __restrict__ y_bf) {
    const int e = blockIdx.x * 256 + threadIdx.x;
    const int b = blockIdx.y;
    __shared__ float sB[16], sC[16];

    float h[Nn];
#pragma unroll
    for (int n = 0; n < Nn; ++n) h[n] = 0.f;
    float Arow[Nn];
#pragma unroll
    for (int n = 0; n < Nn; ++n) Arow[n] = -__expf(A_log[e * Nn + n]);
    const float dpe = Dp[e];

    for (int l = 0; l < Ln; ++l) {
        const size_t bl = (size_t)b * Ln + l;
        if (threadIdx.x < 32) {
            float v = xdbl[bl * 48 + 16 + threadIdx.x];
            if (threadIdx.x < 16) sB[threadIdx.x] = v;
            else                  sC[threadIdx.x - 16] = v;
        }
        __syncthreads();
        const float dtv = dt[bl * En + e];
        const float u   = xc[bl * En + e];
        const float du  = dtv * u;
        float acc = 0.f;
#pragma unroll
        for (int n = 0; n < Nn; ++n) {
            float dA = __expf(dtv * Arow[n]);
            h[n] = dA * h[n] + du * sB[n];
            acc += h[n] * sC[n];
        }
        acc += u * dpe;
        const float z = xz[bl * (2 * En) + En + e];
        y_bf[bl * En + e] = (__bf16)(acc * siluf(z));
        __syncthreads();
    }
}

// ---------------------------------------------------------------------------
// Kernel 8: out_proj GEMM, accumulates:  acc[b,d,l] += sum_e y[b,l,e]*w[d,e]
// ---------------------------------------------------------------------------
__global__ void outproj_kernel(const __bf16* __restrict__ y_bf,
                               const __bf16* __restrict__ w,     // bf16 [D][E]
                               float* __restrict__ acc) {
    const int lane = threadIdx.x;
    const int l0 = blockIdx.x * 16;
    const int d0 = blockIdx.y * 16;
    const int b  = blockIdx.z;

    f32x8 c = {};
    const __bf16* abase = y_bf + ((size_t)b * Ln + l0) * En;
    const __bf16* bbase = w + (size_t)d0 * En;
    for (int kb = 0; kb < En; kb += 32) {
        if (kb + 32 < En) __builtin_prefetch(abase + kb + 32, 0, 3);
        bf16x16 a  = load_a_bf(abase + kb, En, lane);
        bf16x16 bm = load_b_bf(bbase + kb, En, lane);
        c = wmma_bf16(a, bm, c);
    }
    const int n  = lane & 15;
    const int mb = (lane >> 4) << 3;
#pragma unroll
    for (int v = 0; v < 8; ++v) {
        int m = mb + v;
        size_t idx = ((size_t)b * Dn + d0 + n) * Ln + l0 + m;
        acc[idx] += 1.0f /*ALPHA*/ * c[v];
    }
}

// ---------------------------------------------------------------------------
// Kernel 9: out = (x + acc) * mask
// ---------------------------------------------------------------------------
__global__ void final_kernel(const float* __restrict__ x,
                             const float* __restrict__ acc,
                             const float* __restrict__ mask,
                             float* __restrict__ out) {
    const size_t idx = (size_t)blockIdx.x * 256 + threadIdx.x;  // over B*D*L
    if (idx >= (size_t)Bn * Dn * Ln) return;
    const int l = (int)(idx % Ln);
    const int b = (int)(idx / ((size_t)Dn * Ln));
    out[idx] = (x[idx] + acc[idx]) * mask[(size_t)b * Ln + l];
}

// ---------------------------------------------------------------------------
// Host launch
// ---------------------------------------------------------------------------
extern "C" void kernel_launch(void* const* d_in, const int* in_sizes, int n_in,
                              void* d_out, int out_size, void* d_ws, size_t ws_size,
                              hipStream_t stream) {
    (void)in_sizes; (void)n_in; (void)out_size; (void)ws_size;
    const float* x    = (const float*)d_in[0];
    const float* enc  = (const float*)d_in[1];
    const float* mask = (const float*)d_in[2];
    const float* ffw  = (const float*)d_in[3];
    const float* ffb  = (const float*)d_in[4];

    // ---- fp32 scratch ----
    float* ws   = (float*)d_ws;
    float* buf0 = ws;                                   // acc / out0  [B,D,L]
    float* buf2 = buf0 + (size_t)Bn * Dn * Ln;          // xz          [B,L,2E]
    float* buf3 = buf2 + (size_t)Bn * Ln * 2 * En;      // xc f32      [B,L,E]
    float* buf4 = buf3 + (size_t)Bn * Ln * En;          // x_dbl       [B,L,48]
    float* buf5 = buf4 + (size_t)Bn * Ln * 48;          // dt          [B,L,E]
    float* f32_end = buf5 + (size_t)Bn * Ln * En;

    // ---- bf16 scratch (after f32 region; all offsets 32B-aligned) ----
    __bf16* bws    = (__bf16*)f32_end;
    __bf16* ffwT   = bws;                               // [K][D][D]
    __bf16* ipw_bf = ffwT   + (size_t)Dn * Dn * Kn;     // [2E][D]
    __bf16* xpw_bf = ipw_bf + (size_t)2 * En * Dn;      // [48][E]
    __bf16* opw_bf = xpw_bf + (size_t)48 * En;          // [D][E]
    __bf16* xnT    = opw_bf + (size_t)Dn * En;          // [B,L,D]
    __bf16* xc_bf  = xnT    + (size_t)Bn * Ln * Dn;     // [B,L,E]
    __bf16* y_bf   = xc_bf  + (size_t)Bn * Ln * En;     // [B,L,E]

    // 0) weight casts (once)
    cast_ffw_kernel<<<(Dn * Dn * Kn + 255) / 256, 256, 0, stream>>>(ffw, ffwT);

    // 1) conv_ff -> buf0
    conv_ff_kernel<<<dim3(Ln / 16, Dn / 16, Bn), 32, 0, stream>>>(x, ffwT, ffb, buf0);

    // 2) self mamba (src = buf0), then cross mamba (src = encoder_states)
    for (int br = 0; br < 2; ++br) {
        const float* src = (br == 0) ? (const float*)buf0 : enc;
        const int p = 5 + br * 9;
        const float* ipw  = (const float*)d_in[p + 0];
        const float* cw   = (const float*)d_in[p + 1];
        const float* cb   = (const float*)d_in[p + 2];
        const float* xpw  = (const float*)d_in[p + 3];
        const float* dtw  = (const float*)d_in[p + 4];
        const float* dtb  = (const float*)d_in[p + 5];
        const float* alog = (const float*)d_in[p + 6];
        const float* dpf  = (const float*)d_in[p + 7];
        const float* opw  = (const float*)d_in[p + 8];

        cast_kernel<<<(2 * En * Dn + 255) / 256, 256, 0, stream>>>(ipw, ipw_bf, 2 * En * Dn);
        cast_kernel<<<(48 * En + 255) / 256, 256, 0, stream>>>(xpw, xpw_bf, 48 * En);
        cast_kernel<<<(Dn * En + 255) / 256, 256, 0, stream>>>(opw, opw_bf, Dn * En);

        instnorm_kernel<<<Bn * Dn, 256, 0, stream>>>(src, mask, xnT);
        inproj_kernel<<<dim3(Ln / 16, 2 * En / 16, Bn), 32, 0, stream>>>(xnT, ipw_bf, buf2);
        dwconv_silu_kernel<<<(Bn * Ln * En) / 256, 256, 0, stream>>>(buf2, cw, cb, buf3, xc_bf);
        xproj_kernel<<<dim3(Ln / 16, 3, Bn), 32, 0, stream>>>(xc_bf, xpw_bf, buf4);
        dt_kernel<<<(Bn * Ln * En) / 256, 256, 0, stream>>>(buf4, dtw, dtb, buf5);
        scan_kernel<<<dim3(En / 256, Bn), 256, 0, stream>>>(buf3, buf5, buf4, buf2,
                                                            alog, dpf, y_bf);
        outproj_kernel<<<dim3(Ln / 16, Dn / 16, Bn), 32, 0, stream>>>(y_bf, opw_bf, buf0);
    }

    // 3) (x + acc) * mask
    final_kernel<<<(Bn * Dn * Ln) / 256, 256, 0, stream>>>(x, buf0, mask, (float*)d_out);
}
